// Metrics_32830730010848
// MI455X (gfx1250) — compile-verified, matrix operand order checked by reading, not run on hardware
//
#include <hip/hip_runtime.h>
#include <math.h>

#define NUM_TH 101
#define NBINS  102
#define SORT_BLOCK 4096
#define THREADS 256

typedef _Float16 v16h __attribute__((ext_vector_type(16)));
typedef float    v8f  __attribute__((ext_vector_type(8)));

__device__ __forceinline__ float safe_div(float nn, float dd) {
    return dd > 0.0f ? nn / fmaxf(dd, 1.0f) : 0.0f;
}

// ---------------- zero small accumulators ----------------
__global__ void k_zero(unsigned* conf, unsigned* histC, unsigned* histL) {
    int t = threadIdx.x;
    if (t < 4) conf[t] = 0u;
    if (t < NBINS) { histC[t] = 0u; histL[t] = 0u; }
}

// ---------------- pass 1: streaming (bandwidth-bound) ----------------
__global__ __launch_bounds__(THREADS)
void k_pass1(const float2* __restrict__ logits, const int* __restrict__ targets,
             unsigned* __restrict__ rec,
             unsigned* conf, unsigned* histC, unsigned* histL, int n) {
    __shared__ unsigned sConf[4];
    __shared__ unsigned sC[NBINS];
    __shared__ unsigned sL[NBINS];
    int t = threadIdx.x;
    if (t < 4) sConf[t] = 0u;
    if (t < NBINS) { sC[t] = 0u; sL[t] = 0u; }
    __syncthreads();
    int base = blockIdx.x * (THREADS * 8);
    #pragma unroll
    for (int j = 0; j < 8; ++j) {
        int i = base + j * THREADS + t;
        if (i < n) {
            __builtin_prefetch((const void*)(logits + i + THREADS * 8), 0, 1);
            float2 l = logits[i];
            int   tr = targets[i];
            float m  = fmaxf(l.x, l.y);
            float e0 = expf(l.x - m), e1 = expf(l.y - m);
            float p1 = e1 / (e0 + e1);
            int pred = (l.y > l.x) ? 1 : 0;        // argmax(softmax), ties -> 0
            atomicAdd(&sConf[tr * 2 + pred], 1u);
            // smallest k in [0,101] with p1 <= (float)k * 0.01f (exact float compares)
            int k = (int)ceilf(p1 * 100.0f);
            k = max(0, min(k, NUM_TH));
            while (k > 0 && p1 <= (float)(k - 1) * 0.01f) --k;
            while (k < NUM_TH && p1 > (float)k * 0.01f) ++k;
            atomicAdd(&sC[k], 1u);
            atomicAdd(&sL[k], (unsigned)tr);
            // 32-bit sortable record: monotone key in bits [31:1], label in bit 0
            rec[i] = (__float_as_uint(p1) << 1) | (unsigned)tr;
        }
    }
    __syncthreads();
    if (t < 4 && sConf[t]) atomicAdd(&conf[t], sConf[t]);
    if (t < NBINS) {
        if (sC[t]) atomicAdd(&histC[t], sC[t]);
        if (sL[t]) atomicAdd(&histL[t], sL[t]);
    }
}

// ---------------- radix sort: count ----------------
__global__ __launch_bounds__(THREADS)
void k_radix_count(const unsigned* __restrict__ src, unsigned* __restrict__ counts,
                   int n, int numBlocks, int shift) {
    __shared__ unsigned h[256];
    int t = threadIdx.x;
    h[t] = 0u;
    __syncthreads();
    int base = blockIdx.x * SORT_BLOCK;
    #pragma unroll
    for (int j = 0; j < 16; ++j) {
        int i = base + j * THREADS + t;
        if (i < n) atomicAdd(&h[(src[i] >> shift) & 255u], 1u);
    }
    __syncthreads();
    counts[(size_t)t * numBlocks + blockIdx.x] = h[t];
}

// ---------------- radix sort: per-digit row exclusive scan (256 blocks) ----------------
__global__ __launch_bounds__(THREADS)
void k_row_scan(unsigned* counts, unsigned* rowTot, int numBlocks) {
    int d = blockIdx.x;
    unsigned* row = counts + (size_t)d * numBlocks;
    __shared__ unsigned sh[THREADS];
    int t = threadIdx.x;
    unsigned carry = 0u;
    for (int base = 0; base < numBlocks; base += THREADS) {
        int i = base + t;
        unsigned v = (i < numBlocks) ? row[i] : 0u;
        sh[t] = v;
        __syncthreads();
        for (int off = 1; off < THREADS; off <<= 1) {
            unsigned x = (t >= off) ? sh[t - off] : 0u;
            __syncthreads();
            sh[t] += x;
            __syncthreads();
        }
        unsigned incl  = sh[t];
        unsigned total = sh[THREADS - 1];
        if (i < numBlocks) row[i] = carry + (incl - v);
        carry += total;
        __syncthreads();
    }
    if (t == 0) rowTot[d] = carry;
}

// ---------------- radix sort: digit base (1 block) ----------------
__global__ void k_base_scan(const unsigned* rowTot, unsigned* digitBase) {
    __shared__ unsigned sh[256];
    int t = threadIdx.x;
    unsigned v = rowTot[t];
    sh[t] = v;
    __syncthreads();
    for (int off = 1; off < 256; off <<= 1) {
        unsigned x = (t >= off) ? sh[t - off] : 0u;
        __syncthreads();
        sh[t] += x;
        __syncthreads();
    }
    digitBase[t] = sh[t] - v;   // exclusive
}

// ---------------- radix sort: stable scatter (wave32 ballot ranking) ----------------
__global__ __launch_bounds__(THREADS)
void k_radix_scatter(const unsigned* __restrict__ src, unsigned* __restrict__ dst,
                     const unsigned* __restrict__ counts, const unsigned* __restrict__ digitBase,
                     int n, int numBlocks, int shift) {
    __shared__ unsigned waveHist[8][256];
    __shared__ unsigned cursor[8][256];
    __shared__ unsigned gOffs[256];
    int t = threadIdx.x;
    int wave = t >> 5, lane = t & 31;
    #pragma unroll
    for (int w = 0; w < 8; ++w) waveHist[w][t] = 0u;
    gOffs[t] = digitBase[t] + counts[(size_t)t * numBlocks + blockIdx.x];
    __syncthreads();

    int waveBase = blockIdx.x * SORT_BLOCK + wave * 512;
    unsigned recv[16];
    // phase 1: per-wave digit histograms (in element order per wave)
    #pragma unroll
    for (int it = 0; it < 16; ++it) {
        int i = waveBase + it * 32 + lane;
        unsigned r = (i < n) ? src[i] : 0u;
        recv[it] = r;
        if (i < n) atomicAdd(&waveHist[wave][(r >> shift) & 255u], 1u);
    }
    __syncthreads();
    // phase 2: cross-wave exclusive prefix per digit -> running cursors
    {
        unsigned run = 0u;
        #pragma unroll
        for (int w = 0; w < 8; ++w) { cursor[w][t] = run; run += waveHist[w][t]; }
    }
    __syncthreads();
    // phase 3: stable scatter; peers via 8-bit ballot voting
    for (int it = 0; it < 16; ++it) {
        int i = waveBase + it * 32 + lane;
        bool active = (i < n);
        unsigned r = recv[it];
        unsigned d = (r >> shift) & 255u;
        unsigned activeMask = (unsigned)__ballot(active);
        unsigned peers = activeMask;
        #pragma unroll
        for (int b = 0; b < 8; ++b) {
            unsigned bit  = (d >> b) & 1u;
            unsigned vote = (unsigned)__ballot(bit != 0u);
            peers &= bit ? vote : ~vote;
        }
        unsigned ltMask = (1u << lane) - 1u;
        unsigned rank   = (unsigned)__popc(peers & ltMask);
        int leader = __ffs(peers) - 1;
        unsigned baseOld = 0u;
        if (active && lane == leader)
            baseOld = atomicAdd(&cursor[wave][d], (unsigned)__popc(peers));
        baseOld = __shfl(baseOld, leader < 0 ? 0 : leader, 32);
        if (active) dst[gOffs[d] + baseOld + rank] = r;
    }
}

// ---------------- per-block label sums via WMMA ones-reduction ----------------
__global__ __launch_bounds__(THREADS)
void k_label_block_sum(const unsigned* __restrict__ rec, unsigned* __restrict__ blockSums, int n) {
    __shared__ unsigned bsum;
    if (threadIdx.x == 0) bsum = 0u;
    __syncthreads();
    int wave = threadIdx.x >> 5, lane = threadIdx.x & 31;
    int base = blockIdx.x * SORT_BLOCK + wave * 512 + lane * 16;
    v16h b;
    #pragma unroll
    for (int j = 0; j < 16; ++j) {
        int i = base + j;
        unsigned lab = (i < n) ? (rec[i] & 1u) : 0u;
        b[j] = (_Float16)(float)lab;
    }
    v16h a;
    #pragma unroll
    for (int j = 0; j < 16; ++j) a[j] = (_Float16)1.0f;
    v8f c = {};
    // D = ones(16x32) x labels(32x16): every D row = column sums of B.
    c = __builtin_amdgcn_wmma_f32_16x16x32_f16(false, a, false, b, (short)0, c, false, false);
    float val = c[0];
    val += __shfl_xor(val, 1, 32);
    val += __shfl_xor(val, 2, 32);
    val += __shfl_xor(val, 4, 32);
    val += __shfl_xor(val, 8, 32);       // lane 0: sum of this wave's 512 labels (exact in f32)
    if (lane == 0) atomicAdd(&bsum, (unsigned)(val + 0.5f));
    __syncthreads();
    if (threadIdx.x == 0) blockSums[blockIdx.x] = bsum;
}

// ---------------- exclusive scan of block sums (1 block) ----------------
__global__ __launch_bounds__(THREADS)
void k_scan_block_sums(const unsigned* __restrict__ blockSums, unsigned* __restrict__ blockPrefix,
                       int numBlocks) {
    __shared__ unsigned sh[THREADS];
    int t = threadIdx.x;
    unsigned carry = 0u;
    for (int base = 0; base < numBlocks; base += THREADS) {
        int i = base + t;
        unsigned v = (i < numBlocks) ? blockSums[i] : 0u;
        sh[t] = v;
        __syncthreads();
        for (int off = 1; off < THREADS; off <<= 1) {
            unsigned x = (t >= off) ? sh[t - off] : 0u;
            __syncthreads();
            sh[t] += x;
            __syncthreads();
        }
        unsigned incl  = sh[t];
        unsigned total = sh[THREADS - 1];
        if (i < numBlocks) blockPrefix[i] = carry + (incl - v);
        carry += total;
        __syncthreads();
    }
}

// ---------------- AP partial sums ----------------
__global__ __launch_bounds__(THREADS)
void k_ap_partial(const unsigned* __restrict__ rec, const unsigned* __restrict__ blockPrefix,
                  const unsigned* __restrict__ conf, double* __restrict__ apPart, int n) {
    __shared__ unsigned sh[THREADS];
    __shared__ double  dsh[THREADS];
    int t = threadIdx.x;
    int base = blockIdx.x * SORT_BLOCK + t * 16;
    unsigned lab[16]; unsigned s = 0u;
    #pragma unroll
    for (int j = 0; j < 16; ++j) {
        int i = base + j;
        lab[j] = (i < n) ? (rec[i] & 1u) : 0u;
        s += lab[j];
    }
    sh[t] = s;
    __syncthreads();
    for (int off = 1; off < THREADS; off <<= 1) {
        unsigned x = (t >= off) ? sh[t - off] : 0u;
        __syncthreads();
        sh[t] += x;
        __syncthreads();
    }
    unsigned excl = sh[t] - s;
    unsigned P    = conf[2] + conf[3];            // fn + tp = total positives
    unsigned pre  = blockPrefix[blockIdx.x] + excl;
    double acc = 0.0;
    #pragma unroll
    for (int j = 0; j < 16; ++j) {
        int i = base + j;
        if (i < n) {
            if (lab[j]) acc += (double)(P - pre) / (double)(n - i);
            pre += lab[j];
        }
    }
    dsh[t] = acc;
    __syncthreads();
    for (int off = THREADS / 2; off > 0; off >>= 1) {   // fixed-order tree: deterministic
        if (t < off) dsh[t] += dsh[t + off];
        __syncthreads();
    }
    if (t == 0) apPart[blockIdx.x] = dsh[0];
}

// ---------------- finalize: all 414 outputs ----------------
__global__ __launch_bounds__(THREADS)
void k_finalize(const unsigned* conf, const unsigned* histC, const unsigned* histL,
                const float* loss, const double* apPart, int numBlocks,
                float* out, int n) {
    __shared__ float idxP[NUM_TH], fnP[NUM_TH];
    __shared__ double dsh[THREADS];
    int t = threadIdx.x;
    double s = 0.0;
    for (int i = t; i < numBlocks; i += THREADS) s += apPart[i];   // fixed per-thread order
    dsh[t] = s;
    __syncthreads();
    for (int off = THREADS / 2; off > 0; off >>= 1) {
        if (t < off) dsh[t] += dsh[t + off];
        __syncthreads();
    }
    if (t == 0) {
        unsigned rc = 0u, rl = 0u;
        for (int b = 0; b < NUM_TH; ++b) {
            rc += histC[b]; rl += histL[b];
            idxP[b] = (float)rc; fnP[b] = (float)rl;   // counts < 2^24, exact in f32
        }
    }
    __syncthreads();
    float tn = (float)conf[0], fp = (float)conf[1], fn = (float)conf[2], tp = (float)conf[3];
    float nf = (float)n;
    float Pf = fn + tp;
    if (t < NUM_TH) {
        float fn_t  = fnP[t];
        float idx_f = idxP[t];
        float tp_t  = Pf - fn_t;
        float tn_t  = idx_f - fn_t;
        float fp_t  = nf - idx_f - tp_t;
        out[5   + t] = safe_div(tp_t, tp_t + fp_t);  // prec_t
        out[106 + t] = safe_div(tp_t, tp_t + fn_t);  // rec_t
        out[207 + t] = (float)t * 0.01f;             // ths
        out[308 + t] = safe_div(tn_t, tn_t + fp_t);  // spec_t
    }
    if (t == 0) {
        out[0] = tn; out[1] = fp; out[2] = fn; out[3] = tp;
        out[4] = (float)(dsh[0] / (double)fmaxf(Pf, 1.0f));     // map_ == ap
        out[409] = safe_div(tp, tp + fn);   // recall
        out[410] = safe_div(tp, tp + fp);   // precision
        out[411] = safe_div(tn, tn + fp);   // specificity
        out[412] = (tp + tn) / nf;          // accuracy
        out[413] = (loss[0] + loss[1] + loss[2] + loss[3]) * 0.25f;
    }
}

extern "C" void kernel_launch(void* const* d_in, const int* in_sizes, int n_in,
                              void* d_out, int out_size, void* d_ws, size_t ws_size,
                              hipStream_t stream) {
    const float2* logits  = (const float2*)d_in[0];
    const int*    targets = (const int*)d_in[1];
    const float*  loss    = (const float*)d_in[2];
    float* out = (float*)d_out;
    int n = in_sizes[1];
    int sortBlocks = (n + SORT_BLOCK - 1) / SORT_BLOCK;

    char* w = (char*)d_ws;
    auto alloc = [&](size_t bytes) -> void* {
        void* p = (void*)w;
        w += (bytes + 255) & ~(size_t)255;
        return p;
    };
    unsigned* recA        = (unsigned*)alloc((size_t)n * 4);
    unsigned* recB        = (unsigned*)alloc((size_t)n * 4);
    unsigned* counts      = (unsigned*)alloc((size_t)256 * sortBlocks * 4);
    unsigned* rowTot      = (unsigned*)alloc(256 * 4);
    unsigned* digitBase   = (unsigned*)alloc(256 * 4);
    unsigned* conf        = (unsigned*)alloc(4 * 4);
    unsigned* histC       = (unsigned*)alloc(NBINS * 4);
    unsigned* histL       = (unsigned*)alloc(NBINS * 4);
    unsigned* blockSums   = (unsigned*)alloc((size_t)sortBlocks * 4);
    unsigned* blockPrefix = (unsigned*)alloc((size_t)sortBlocks * 4);
    double*   apPart      = (double*)alloc((size_t)sortBlocks * 8);

    k_zero<<<1, 256, 0, stream>>>(conf, histC, histL);
    int p1Blocks = (n + THREADS * 8 - 1) / (THREADS * 8);
    k_pass1<<<p1Blocks, THREADS, 0, stream>>>(logits, targets, recA, conf, histC, histL, n);

    unsigned* src = recA; unsigned* dst = recB;
    for (int pass = 0; pass < 4; ++pass) {
        int shift = pass * 8;
        k_radix_count  <<<sortBlocks, THREADS, 0, stream>>>(src, counts, n, sortBlocks, shift);
        k_row_scan     <<<256,        THREADS, 0, stream>>>(counts, rowTot, sortBlocks);
        k_base_scan    <<<1,          256,     0, stream>>>(rowTot, digitBase);
        k_radix_scatter<<<sortBlocks, THREADS, 0, stream>>>(src, dst, counts, digitBase, n, sortBlocks, shift);
        unsigned* tmp = src; src = dst; dst = tmp;
    }
    // after 4 passes sorted data is back in recA (== src)
    k_label_block_sum<<<sortBlocks, THREADS, 0, stream>>>(src, blockSums, n);
    k_scan_block_sums<<<1,          THREADS, 0, stream>>>(blockSums, blockPrefix, sortBlocks);
    k_ap_partial     <<<sortBlocks, THREADS, 0, stream>>>(src, blockPrefix, conf, apPart, n);
    k_finalize       <<<1,          THREADS, 0, stream>>>(conf, histC, histL, loss, apPart, sortBlocks, out, n);
    (void)n_in; (void)out_size; (void)ws_size;
}
